// Specformer_22333829939746
// MI455X (gfx1250) — matrix operand order repaired and checked by reading, not compiled, not run on hardware
//
#include <hip/hip_runtime.h>
#include <hip/hip_bf16.h>
#include <math.h>

// ---- problem constants ----
#define BB   4
#define NN   16384
#define KK   128     // fourier modes
#define DD   128     // model dim
#define HH   8
#define LL   6
#define DFFN 512
#define NFF  16
#define DCC  64
#define MM   257     // spectral tokens (2K+1)
#define MP   288     // padded to multiple of 32 for WMMA K-chunks
#define DH   16

typedef __bf16 v16bf __attribute__((ext_vector_type(16)));
typedef float  v8f   __attribute__((ext_vector_type(8)));

// ------------------------------------------------------------------
// WMMA helper (CDNA5 wave32, V_WMMA_F32_16X16X32_BF16, f32 accum)
// ------------------------------------------------------------------
__device__ __forceinline__ v8f wmma_bf16(v16bf a, v16bf b, v8f c) {
  return __builtin_amdgcn_wmma_f32_16x16x32_bf16(
      false, a, false, b, (short)0, c, false, false);
}

// A 16x32 fragment from memory where K is innermost (rowStride between M-rows).
// lanes 0-15: row=lane,    K = 0..7  then 16..23
// lanes 16-31: row=lane-16, K = 8..15 then 24..31
__device__ __forceinline__ v16bf frag_a(const __bf16* base, size_t rowStride) {
  int lane = threadIdx.x & 31;
  int r  = lane & 15;
  int kb = (lane >> 4) << 3;
  const __bf16* p = base + (size_t)r * rowStride + kb;
  union { uint4 q[2]; v16bf v; } u;
  u.q[0] = *(const uint4*)p;
  u.q[1] = *(const uint4*)(p + 16);
  return u.v;
}

// B 32x16 fragment from K-innermost (transposed) memory [n][k].
// lanes 0-15: col=lane, K=0..7 then 16..23 ; lanes 16-31: col=lane-16, K=8..15 then 24..31
__device__ __forceinline__ v16bf frag_b(const __bf16* base, size_t colStride) {
  int lane = threadIdx.x & 31;
  int c  = lane & 15;
  int kb = (lane >> 4) << 3;
  const __bf16* p = base + (size_t)c * colStride + kb;
  union { uint4 q[2]; v16bf v; } u;
  u.q[0] = *(const uint4*)p;
  u.q[1] = *(const uint4*)(p + 16);
  return u.v;
}

// block reduction over 128 threads (4 waves)
__device__ __forceinline__ float block_reduce128(float v, volatile float* sc4) {
#pragma unroll
  for (int off = 16; off >= 1; off >>= 1) v += __shfl_xor(v, off, 32);
  __syncthreads();
  if ((threadIdx.x & 31) == 0) sc4[threadIdx.x >> 5] = v;
  __syncthreads();
  return sc4[0] + sc4[1] + sc4[2] + sc4[3];
}

// ------------------------------------------------------------------
// K0: f32 [L][R][C] -> bf16 transposed [L][C][R]  (one-time weights)
// ------------------------------------------------------------------
__global__ void k_cvt_transpose(const float* __restrict__ src, __bf16* __restrict__ dst,
                                int R, int C, int total) {
  int i = blockIdx.x * blockDim.x + threadIdx.x;
  if (i >= total) return;
  int l = i / (R * C);
  int rem = i - l * (R * C);
  int r = rem / C, c = rem - r * C;
  dst[((size_t)l * C + c) * R + r] = (__bf16)src[i];
}

// ------------------------------------------------------------------
// K1: bases (row-major), weighted-bases (transposed), coord-embed, input proj
//     outputs: basesRM[b][x][m], wbT[b][m][x], h[b][x][d], hT[b][d][x]
// Weights are wave-uniform -> read directly (compiler scalarizes to s_load).
// ------------------------------------------------------------------
__global__ __launch_bounds__(256) void k_build(
    const float* __restrict__ u, const float* __restrict__ nodes,
    const float* __restrict__ nw, const float* __restrict__ modes,
    const float* __restrict__ coordB, const float* __restrict__ cp_w,
    const float* __restrict__ cp_b, const float* __restrict__ in_w,
    const float* __restrict__ in_b,
    __bf16* __restrict__ basesRM, __bf16* __restrict__ wbT,
    float* __restrict__ h, __bf16* __restrict__ hT)
{
  int tid = threadIdx.x;
  size_t p = (size_t)blockIdx.x * 256 + tid;   // 0 .. B*N-1
  size_t b = p >> 14;                          // N = 2^14
  size_t x = p & (NN - 1);
  float nx = nodes[p * 2], ny = nodes[p * 2 + 1];
  float w  = nw[p];
  __bf16* bas  = basesRM + p * MP;
  __bf16* wcol = wbT + (size_t)b * MP * NN + x;
  for (int k = 0; k < KK; ++k) {
    float t  = nx * modes[k * 2] + ny * modes[k * 2 + 1];
    float cv = __cosf(t), sv = __sinf(t);
    bas[k]      = (__bf16)cv;  bas[KK + k] = (__bf16)sv;
    wcol[(size_t)k * NN]        = (__bf16)(cv * w);
    wcol[(size_t)(KK + k) * NN] = (__bf16)(sv * w);
  }
  bas[2 * KK] = (__bf16)1.0f;
  wcol[(size_t)(2 * KK) * NN] = (__bf16)w;
  for (int m = 2 * KK + 1; m < MP; ++m) {
    bas[m] = (__bf16)0.0f;
    wcol[(size_t)m * NN] = (__bf16)0.0f;
  }

  // coord embedding
  float f[34];
  f[0] = nx; f[1] = ny;
#pragma unroll
  for (int j = 0; j < NFF; ++j) {
    float xb = nx * coordB[j] + ny * coordB[NFF + j];
    f[2 + j]       = __sinf(xb);
    f[2 + NFF + j] = __cosf(xb);
  }
  float c[DCC];
#pragma unroll
  for (int j = 0; j < DCC; ++j) c[j] = cp_b[j];
  for (int i = 0; i < 34; ++i) {
    float fi = f[i];
#pragma unroll
    for (int j = 0; j < DCC; ++j) c[j] += fi * cp_w[i * DCC + j];
  }
  float u0 = u[p * 2], u1 = u[p * 2 + 1];
  float* hp = h + p * DD;
  for (int d = 0; d < DD; ++d) {
    float acc = in_b[d] + u0 * in_w[d] + u1 * in_w[DD + d];
#pragma unroll
    for (int i = 0; i < DCC; ++i) acc += c[i] * in_w[(2 + i) * DD + d];
    hp[d] = acc;
    hT[((size_t)b * DD + d) * NN + x] = (__bf16)acc;
  }
}

// ------------------------------------------------------------------
// K2: slice  Z[b,m,d] = sum_x wbases[x,m] * h[x,d]    (WMMA, no LDS)
// A = wbT[m][x], B = hT[d][x]; grid (MP/16, B), 256 thr, wave = d-tile
// Two independent accumulator chains to pipeline the XDL.
// ------------------------------------------------------------------
__global__ __launch_bounds__(256) void k_slice(
    const __bf16* __restrict__ wbT, const __bf16* __restrict__ hT,
    float* __restrict__ Z)
{
  int tid = threadIdx.x, wave = tid >> 5, lane = tid & 31;
  int m0 = blockIdx.x * 16, b = blockIdx.y;
  const __bf16* Abase = wbT + ((size_t)b * MP + m0) * NN;
  const __bf16* Bbase = hT  + ((size_t)b * DD + wave * 16) * NN;
  v8f acc0 = {0.f, 0.f, 0.f, 0.f, 0.f, 0.f, 0.f, 0.f};
  v8f acc1 = acc0;
  for (int xg = 0; xg < NN; xg += 256) {       // group of 8 K-chunks
    // pull next group's lines toward the WGP while WMMAs run
    __builtin_prefetch((const void*)(Abase + xg + 256), 0, 3);
    __builtin_prefetch((const void*)(Bbase + xg + 256), 0, 3);
#pragma unroll
    for (int xo = 0; xo < 256; xo += 64) {
      int x0 = xg + xo;
      acc0 = wmma_bf16(frag_a(Abase + x0,      NN), frag_b(Bbase + x0,      NN), acc0);
      acc1 = wmma_bf16(frag_a(Abase + x0 + 32, NN), frag_b(Bbase + x0 + 32, NN), acc1);
    }
  }
  v8f acc = acc0 + acc1;
  int col = wave * 16 + (lane & 15);
  int rb  = (lane >> 4) << 3;
#pragma unroll
  for (int v = 0; v < 8; ++v)
    Z[((size_t)b * MP + m0 + rb + v) * DD + col] = acc[v];
}

// ------------------------------------------------------------------
// K3: LN1 + QKV projection. grid (M, B), 128 thr
// ------------------------------------------------------------------
__global__ __launch_bounds__(128) void k_ln1_qkv(
    const float* __restrict__ Z, const float* __restrict__ ln_w,
    const float* __restrict__ ln_b, const float* __restrict__ qkv_w,
    const float* __restrict__ qkv_b, float* __restrict__ qkv, int l)
{
  __shared__ float zin[DD];
  __shared__ float sc[4];
  int m = blockIdx.x, b = blockIdx.y, tid = threadIdx.x;
  float z  = Z[((size_t)b * MP + m) * DD + tid];
  float s  = block_reduce128(z, sc);
  float mean = s * (1.0f / DD);
  float dv = z - mean;
  float s2 = block_reduce128(dv * dv, sc);
  float inv = rsqrtf(s2 * (1.0f / DD) + 1e-5f);
  zin[tid] = dv * inv * ln_w[l * DD + tid] + ln_b[l * DD + tid];
  __syncthreads();
  const float* W = qkv_w + (size_t)l * DD * 3 * DD;
  float* out = qkv + ((size_t)b * MM + m) * (3 * DD);
#pragma unroll
  for (int jj = 0; jj < 3; ++jj) {
    int j = tid + jj * DD;
    float acc = qkv_b[l * 3 * DD + j];
    for (int i = 0; i < DD; ++i) acc += zin[i] * W[i * 3 * DD + j];
    out[j] = acc;
  }
}

// ------------------------------------------------------------------
// K4: attention core, online softmax. grid (H, B), 256 thr
// ------------------------------------------------------------------
__global__ __launch_bounds__(256) void k_attn(
    const float* __restrict__ qkv, float* __restrict__ o)
{
  __shared__ float ks[MM * DH];
  __shared__ float vs[MM * DH];
  int hh = blockIdx.x, b = blockIdx.y, tid = threadIdx.x;
  const float* q0 = qkv + (size_t)b * MM * (3 * DD);
  for (int i = tid; i < MM * DH; i += 256) {
    int m = i >> 4, e = i & 15;
    ks[i] = q0[(size_t)m * (3 * DD) + DD     + hh * DH + e];
    vs[i] = q0[(size_t)m * (3 * DD) + 2 * DD + hh * DH + e];
  }
  __syncthreads();
  for (int r = tid; r < MM; r += 256) {
    float q[DH];
#pragma unroll
    for (int e = 0; e < DH; ++e) q[e] = q0[(size_t)r * (3 * DD) + hh * DH + e];
    float mx = -3.0e38f, ssum = 0.f;
    float oacc[DH];
#pragma unroll
    for (int e = 0; e < DH; ++e) oacc[e] = 0.f;
    for (int m = 0; m < MM; ++m) {
      float s = 0.f;
#pragma unroll
      for (int e = 0; e < DH; ++e) s += q[e] * ks[m * DH + e];
      s *= 0.25f;                                   // 1/sqrt(DH)
      float nm   = fmaxf(mx, s);
      float corr = __expf(mx - nm);
      float pv   = __expf(s - nm);
      ssum = ssum * corr + pv;
#pragma unroll
      for (int e = 0; e < DH; ++e) oacc[e] = oacc[e] * corr + pv * vs[m * DH + e];
      mx = nm;
    }
    float invs = 1.0f / ssum;
    float* op = o + ((size_t)b * MM + r) * DD + hh * DH;
#pragma unroll
    for (int e = 0; e < DH; ++e) op[e] = oacc[e] * invs;
  }
}

// ------------------------------------------------------------------
// K5: attn out projection + residual into Z; emit Z transposed bf16 [d][m].
// grid (MP/16, B), 256 thr ; thread = (d, half) computing 8 m-rows
// ------------------------------------------------------------------
__global__ __launch_bounds__(256) void k_ao(
    const float* __restrict__ o, const float* __restrict__ ao_w,
    const float* __restrict__ ao_b, float* __restrict__ Z,
    __bf16* __restrict__ ZT, int l)
{
  __shared__ float osh[16 * DD];
  int tid = threadIdx.x;
  int m0 = blockIdx.x * 16, b = blockIdx.y;
  for (int i = tid; i < 16 * DD; i += 256) {
    int m = i >> 7, d = i & 127;
    int mg = m0 + m;
    osh[i] = (mg < MM) ? o[((size_t)b * MM + mg) * DD + d] : 0.f;
  }
  __syncthreads();
  int d = tid & 127, mgrp = tid >> 7;      // 0 or 1
  const float* W = ao_w + (size_t)l * DD * DD;
  float acc[8];
#pragma unroll
  for (int j = 0; j < 8; ++j) acc[j] = 0.f;
  for (int i = 0; i < DD; ++i) {
    float wv = W[i * DD + d];
#pragma unroll
    for (int j = 0; j < 8; ++j) acc[j] += osh[(mgrp * 8 + j) * DD + i] * wv;
  }
  float bias = ao_b[l * DD + d];
  alignas(16) __bf16 z8[8];
#pragma unroll
  for (int j = 0; j < 8; ++j) {
    int m = m0 + mgrp * 8 + j;
    size_t zi = ((size_t)b * MP + m) * DD + d;
    float z = Z[zi];
    if (m < MM) { z += acc[j] + bias; Z[zi] = z; }
    z8[j] = (__bf16)z;
  }
  *((uint4*)(ZT + ((size_t)b * DD + d) * MP + m0 + mgrp * 8)) = *((const uint4*)z8);
}

// ------------------------------------------------------------------
// K6: deslice h += bases @ Z  (WMMA, no staging) fused with LN2 -> hn bf16
// A = basesRM[x][m] (K=m innermost), B = ZT[d][m]; grid (N/16, B), 256 thr
// ------------------------------------------------------------------
__global__ __launch_bounds__(256) void k_deslice(
    const __bf16* __restrict__ basesRM, const __bf16* __restrict__ ZT,
    float* __restrict__ h, const float* __restrict__ ln_w,
    const float* __restrict__ ln_b, __bf16* __restrict__ hnbf, int l)
{
  __shared__ float hout[16 * DD];
  int tid = threadIdx.x, wave = tid >> 5, lane = tid & 31;
  int x0 = blockIdx.x * 16, b = blockIdx.y;
  const __bf16* Abase = basesRM + ((size_t)b * NN + x0) * MP;
  const __bf16* Bbase = ZT + ((size_t)b * DD + wave * 16) * MP;
  v8f acc0 = {0.f, 0.f, 0.f, 0.f, 0.f, 0.f, 0.f, 0.f};
  v8f acc1 = acc0;
#pragma unroll
  for (int kc = 0; kc < 256; kc += 64) {     // 8 chunks, two chains
    acc0 = wmma_bf16(frag_a(Abase + kc,      MP), frag_b(Bbase + kc,      MP), acc0);
    acc1 = wmma_bf16(frag_a(Abase + kc + 32, MP), frag_b(Bbase + kc + 32, MP), acc1);
  }
  acc0 = wmma_bf16(frag_a(Abase + 256, MP), frag_b(Bbase + 256, MP), acc0); // tail chunk
  v8f acc = acc0 + acc1;

  int col = wave * 16 + (lane & 15);
  int rb  = (lane >> 4) << 3;
#pragma unroll
  for (int v = 0; v < 8; ++v) {
    int row = rb + v;
    size_t hi = ((size_t)b * NN + x0 + row) * DD + col;
    float hv = h[hi] + acc[v];
    h[hi] = hv;
    hout[row * DD + col] = hv;
  }
  __syncthreads();

  // LN2: 16 rows x 16 slots x 8 elems, shuffle reduce within 16 lanes
  int row = tid >> 4, slot = tid & 15;
  float s = 0.f, s2 = 0.f;
  float vals[8];
#pragma unroll
  for (int j = 0; j < 8; ++j) {
    float v = hout[row * DD + slot * 8 + j];
    vals[j] = v; s += v; s2 += v * v;
  }
#pragma unroll
  for (int off = 8; off >= 1; off >>= 1) {
    s  += __shfl_xor(s, off, 32);
    s2 += __shfl_xor(s2, off, 32);
  }
  float mean = s * (1.0f / DD);
  float var  = s2 * (1.0f / DD) - mean * mean;
  float inv  = rsqrtf(var + 1e-5f);
  alignas(16) __bf16 hn8[8];
#pragma unroll
  for (int j = 0; j < 8; ++j) {
    int dd = slot * 8 + j;
    hn8[j] = (__bf16)((vals[j] - mean) * inv * ln_w[l * DD + dd] + ln_b[l * DD + dd]);
  }
  *((uint4*)(hnbf + ((size_t)b * NN + x0 + row) * DD + slot * 8)) = *((const uint4*)hn8);
}

// ------------------------------------------------------------------
// K7: FFN  h += gelu(hn @ f1 + b1) @ f2 + b2 ; also emits hT for next slice
// A1 = hn[x][k] row-major; B1 = w1T[n][k]; B2 = w2T[d][k]; act via LDS
// grid (N/16, B), 256 thr
// ------------------------------------------------------------------
__global__ __launch_bounds__(256) void k_ffn(
    const __bf16* __restrict__ hnbf, const __bf16* __restrict__ w1T,
    const float* __restrict__ b1, const __bf16* __restrict__ w2T,
    const float* __restrict__ b2, float* __restrict__ h,
    __bf16* __restrict__ hT, int l)
{
  __shared__ alignas(16) __bf16 sAct[16 * DFFN];  // 16 KB
  __shared__ float hsc[16 * 132];                 // padded stride, 8.25 KB
  int tid = threadIdx.x, wave = tid >> 5, lane = tid & 31;
  int x0 = blockIdx.x * 16, b = blockIdx.y;

  // ---- GEMM1: [16,128] x [128,512] ---- (4 independent chains over j)
  const __bf16* Ab = hnbf + ((size_t)b * NN + x0) * DD;
  const __bf16* W1 = w1T + (size_t)l * DFFN * DD;
  v8f acc[4];
#pragma unroll
  for (int j = 0; j < 4; ++j) acc[j] = (v8f){0.f,0.f,0.f,0.f,0.f,0.f,0.f,0.f};
#pragma unroll
  for (int kc = 0; kc < 4; ++kc) {
    v16bf a = frag_a(Ab + kc * 32, DD);
#pragma unroll
    for (int j = 0; j < 4; ++j) {
      int n0 = (wave * 4 + j) * 16;
      v16bf bb = frag_b(W1 + (size_t)n0 * DD + kc * 32, DD);
      acc[j] = wmma_bf16(a, bb, acc[j]);
    }
  }
  // bias + exact GELU -> sAct
  int rb = (lane >> 4) << 3;
#pragma unroll
  for (int j = 0; j < 4; ++j) {
    int col = (wave * 4 + j) * 16 + (lane & 15);
    float bj = b1[l * DFFN + col];
#pragma unroll
    for (int v = 0; v < 8; ++v) {
      float xg = acc[j][v] + bj;
      float g = 0.5f * xg * (1.0f + erff(xg * 0.70710678118f));
      sAct[(rb + v) * DFFN + col] = (__bf16)g;
    }
  }
  __syncthreads();

  // ---- GEMM2: [16,512] x [512,128] ---- (two chains)
  const __bf16* W2 = w2T + (size_t)l * DD * DFFN;
  v8f acc2a = {0.f, 0.f, 0.f, 0.f, 0.f, 0.f, 0.f, 0.f};
  v8f acc2b = acc2a;
#pragma unroll
  for (int kc = 0; kc < 16; kc += 2) {
    acc2a = wmma_bf16(frag_a(sAct + kc * 32, DFFN),
                      frag_b(W2 + (size_t)(wave * 16) * DFFN + kc * 32, DFFN), acc2a);
    acc2b = wmma_bf16(frag_a(sAct + kc * 32 + 32, DFFN),
                      frag_b(W2 + (size_t)(wave * 16) * DFFN + kc * 32 + 32, DFFN), acc2b);
  }
  v8f acc2 = acc2a + acc2b;

  int col = wave * 16 + (lane & 15);
  float bd = b2[l * DD + col];
#pragma unroll
  for (int v = 0; v < 8; ++v) {
    size_t hi = ((size_t)b * NN + x0 + rb + v) * DD + col;
    float hv = h[hi] + acc2[v] + bd;
    h[hi] = hv;
    hsc[(rb + v) * 132 + col] = hv;
  }
  __syncthreads();
  // transpose tile -> hT[d][x] as packed bf16 (bank-safe reads, 32B stores)
  if (tid < DD) {
    alignas(16) __bf16 t16[16];
#pragma unroll
    for (int xx = 0; xx < 16; ++xx) t16[xx] = (__bf16)hsc[xx * 132 + tid];
    uint4* dst = (uint4*)(hT + ((size_t)b * DD + tid) * NN + x0);
    dst[0] = ((const uint4*)t16)[0];
    dst[1] = ((const uint4*)t16)[1];
  }
}

// ------------------------------------------------------------------
// K8: final LN + output projection [128 -> 2]
// ------------------------------------------------------------------
__global__ __launch_bounds__(128) void k_final(
    const float* __restrict__ h, const float* __restrict__ ln_w,
    const float* __restrict__ ln_b, const float* __restrict__ out_w,
    const float* __restrict__ out_b, float* __restrict__ out)
{
  __shared__ float sc[4];
  size_t p = blockIdx.x;
  int tid = threadIdx.x;
  float v  = h[p * DD + tid];
  float s  = block_reduce128(v, sc);
  float mean = s * (1.0f / DD);
  float dv = v - mean;
  float s2 = block_reduce128(dv * dv, sc);
  float inv = rsqrtf(s2 * (1.0f / DD) + 1e-5f);
  float hn  = dv * inv * ln_w[tid] + ln_b[tid];
  float o0 = block_reduce128(hn * out_w[tid * 2], sc);
  float o1 = block_reduce128(hn * out_w[tid * 2 + 1], sc);
  if (tid == 0) {
    out[p * 2]     = o0 + out_b[0];
    out[p * 2 + 1] = o1 + out_b[1];
  }
}

// ------------------------------------------------------------------
extern "C" void kernel_launch(void* const* d_in, const int* in_sizes, int n_in,
                              void* d_out, int out_size, void* d_ws, size_t ws_size,
                              hipStream_t stream) {
  const float* u      = (const float*)d_in[0];
  const float* nodes  = (const float*)d_in[1];
  const float* nw     = (const float*)d_in[2];
  const float* modes  = (const float*)d_in[3];
  const float* coordB = (const float*)d_in[4];
  const float* cp_w   = (const float*)d_in[5];
  const float* cp_b   = (const float*)d_in[6];
  const float* in_w   = (const float*)d_in[7];
  const float* in_b   = (const float*)d_in[8];
  const float* ln1_w  = (const float*)d_in[9];
  const float* ln1_b  = (const float*)d_in[10];
  const float* qkv_w  = (const float*)d_in[11];
  const float* qkv_b  = (const float*)d_in[12];
  const float* ao_w   = (const float*)d_in[13];
  const float* ao_b   = (const float*)d_in[14];
  const float* f1_w   = (const float*)d_in[15];
  const float* f1_b   = (const float*)d_in[16];
  const float* f2_w   = (const float*)d_in[17];
  const float* f2_b   = (const float*)d_in[18];
  const float* ln2_w  = (const float*)d_in[19];
  const float* ln2_b  = (const float*)d_in[20];
  const float* lnf_w  = (const float*)d_in[21];
  const float* lnf_b  = (const float*)d_in[22];
  const float* out_w  = (const float*)d_in[23];
  const float* out_b  = (const float*)d_in[24];

  // workspace carve-up (256B aligned)
  char* ws = (char*)d_ws;
  size_t off = 0;
  auto alloc = [&](size_t bytes) -> char* {
    char* p = ws + off;
    off = (off + bytes + 255) & ~(size_t)255;
    return p;
  };
  __bf16* basesRM = (__bf16*)alloc((size_t)BB * NN * MP * 2);   // [b][x][m]
  __bf16* wbT     = (__bf16*)alloc((size_t)BB * MP * NN * 2);   // [b][m][x]
  __bf16* hTb     = (__bf16*)alloc((size_t)BB * DD * NN * 2);   // [b][d][x]
  float*  hf      = (float* )alloc((size_t)BB * NN * DD * 4);   // [b][x][d]
  __bf16* hnbfb   = (__bf16*)alloc((size_t)BB * NN * DD * 2);   // [b][x][d]
  float*  Zf      = (float* )alloc((size_t)BB * MP * DD * 4);   // [b][m][d]
  __bf16* ZTb     = (__bf16*)alloc((size_t)BB * DD * MP * 2);   // [b][d][m]
  float*  qkvbuf  = (float* )alloc((size_t)BB * MM * 3 * DD * 4);
  float*  obuf    = (float* )alloc((size_t)BB * MM * DD * 4);
  __bf16* w1Tb    = (__bf16*)alloc((size_t)LL * DFFN * DD * 2); // [l][n][k]
  __bf16* w2Tb    = (__bf16*)alloc((size_t)LL * DD * DFFN * 2); // [l][d][k]

  // one-time prep: transpose weights to K-innermost bf16
  int nwt = LL * DD * DFFN;
  k_cvt_transpose<<<(nwt + 255) / 256, 256, 0, stream>>>(f1_w, w1Tb, DD, DFFN, nwt);
  k_cvt_transpose<<<(nwt + 255) / 256, 256, 0, stream>>>(f2_w, w2Tb, DFFN, DD, nwt);
  k_build<<<BB * NN / 256, 256, 0, stream>>>(u, nodes, nw, modes, coordB,
                                             cp_w, cp_b, in_w, in_b,
                                             basesRM, wbT, hf, hTb);

  for (int l = 0; l < LL; ++l) {
    k_slice  <<<dim3(MP / 16, BB), 256, 0, stream>>>(wbT, hTb, Zf);
    k_ln1_qkv<<<dim3(MM, BB),      128, 0, stream>>>(Zf, ln1_w, ln1_b, qkv_w, qkv_b, qkvbuf, l);
    k_attn   <<<dim3(HH, BB),      256, 0, stream>>>(qkvbuf, obuf);
    k_ao     <<<dim3(MP / 16, BB), 256, 0, stream>>>(obuf, ao_w, ao_b, Zf, ZTb, l);
    k_deslice<<<dim3(NN / 16, BB), 256, 0, stream>>>(basesRM, ZTb, hf, ln2_w, ln2_b, hnbfb, l);
    k_ffn    <<<dim3(NN / 16, BB), 256, 0, stream>>>(hnbfb, w1Tb, f1_b, w2Tb, f2_b, hf, hTb, l);
  }
  k_final<<<BB * NN, 128, 0, stream>>>(hf, lnf_w, lnf_b, out_w, out_b, (float*)d_out);
}
